// LongformerAttention_43825846288475
// MI455X (gfx1250) — matrix operand order repaired and checked by reading, not compiled, not run on hardware
//
#include <hip/hip_runtime.h>

// ---------------------------------------------------------------------------
// Longformer attention block for gfx1250 (MI455X), bf16 WMMA + f32 accumulate.
// B=2, S=2048, D=1024, H=16, d=64, window=256, global token {0}.
// ---------------------------------------------------------------------------

typedef __attribute__((ext_vector_type(16))) __bf16 v16bf;
typedef __attribute__((ext_vector_type(8)))  __bf16 v8bf;
typedef __attribute__((ext_vector_type(8)))  float  v8f;

union AFrag {                  // 16x32 bf16 A-matrix fragment (8 VGPRs/lane)
    v16bf v;
    struct { v8bf lo; v8bf hi; } p;
};

__device__ __forceinline__ __bf16 f2bf(float f) {
    unsigned u = __builtin_bit_cast(unsigned, f);
    u += 0x7FFFu + ((u >> 16) & 1u);               // round-to-nearest-even
    unsigned short h = (unsigned short)(u >> 16);
    return __builtin_bit_cast(__bf16, h);
}

__device__ __forceinline__ v8f wmma_bf16(v16bf a, v16bf b, v8f c) {
    // D(f32 16x16) = A(bf16 16x32) * B(bf16 32x16) + C
    return __builtin_amdgcn_wmma_f32_16x16x32_bf16(false, a, false, b,
                                                   (short)0, c, false, false);
}

// XOR shuffle inside 16-lane groups via ds_swizzle (group-of-32 encoding:
// offset = xor<<10 | or<<5 | and ; and=0x1f full sharing, or=0).
template <int M>
__device__ __forceinline__ float swz(float v) {
    return __builtin_bit_cast(
        float, __builtin_amdgcn_ds_swizzle(__builtin_bit_cast(int, v),
                                           (M << 10) | 0x1f));
}

// ---------------------------------------------------------------------------
// Kernel 1: f32 -> bf16 pack of x  (4096*1024 elements)
// ---------------------------------------------------------------------------
__global__ void lf_cvt_x(const float* __restrict__ x, __bf16* __restrict__ xb) {
    int i = blockIdx.x * 256 + threadIdx.x;
    xb[i] = f2bf(x[i]);
}

// ---------------------------------------------------------------------------
// Kernel 2: W[K=1024][N=1024] f32 -> Wt[N][K] bf16 (transpose + pack)
// ---------------------------------------------------------------------------
__global__ void lf_transpose_w(const float* __restrict__ W, __bf16* __restrict__ Wt) {
    int i = blockIdx.x * 256 + threadIdx.x;   // linear over Wt (n major, k minor)
    int n = i >> 10;
    int k = i & 1023;
    Wt[i] = f2bf(W[(k << 10) + n]);
}

// ---------------------------------------------------------------------------
// Kernel 3: generic GEMM  C[4096,1024] = A_bf16[4096,1024] @ Bt_bf16[1024,1024]^T + bias
// One wave computes a 16x64 tile with 4x v_wmma_f32_16x16x32_bf16 per K-chunk.
// mode 0: write Q   bf16 [B*H, S, d], scaled by 1/sqrt(64)
// mode 1: write K   bf16 [B*H, S, d]
// mode 2: write V^T bf16 [B*H, d, S]
// mode 3: write f32 row-major [4096,1024] (final output)
// ---------------------------------------------------------------------------
__global__ void lf_gemm(const __bf16* __restrict__ A, const __bf16* __restrict__ Bt,
                        const float* __restrict__ bias, void* __restrict__ outp,
                        int mode) {
    const int lane = threadIdx.x & 31;
    const int gw   = (blockIdx.x * blockDim.x + threadIdx.x) >> 5;   // 0..4095
    const int n0   = (gw & 15) << 6;    // 64-wide N tile
    const int m0   = (gw >> 4) << 4;    // 16-row  M tile
    const int ml   = lane & 15;
    const int kh   = lane >> 4;

    const __bf16* arow = A + (size_t)(m0 + ml) * 1024;

    const v8f zf = {0.f, 0.f, 0.f, 0.f, 0.f, 0.f, 0.f, 0.f};
    v8f acc[4] = {zf, zf, zf, zf};

    for (int k0 = 0; k0 < 1024; k0 += 32) {
        AFrag a;
        a.p.lo = *reinterpret_cast<const v8bf*>(arow + k0 + kh * 16);
        a.p.hi = *reinterpret_cast<const v8bf*>(arow + k0 + kh * 16 + 8);
        __builtin_prefetch(arow + k0 + 32, 0, 0);   // global_prefetch_b8
#pragma unroll
        for (int j = 0; j < 4; ++j) {
            const v16bf bfr = *reinterpret_cast<const v16bf*>(
                Bt + (size_t)(n0 + j * 16 + ml) * 1024 + k0 + kh * 16);
            acc[j] = wmma_bf16(a.v, bfr, acc[j]);
        }
    }

#pragma unroll
    for (int j = 0; j < 4; ++j) {
        const int col = n0 + j * 16 + ml;
        const float bj = bias[col];
#pragma unroll
        for (int r = 0; r < 8; ++r) {
            const int m = m0 + kh * 8 + r;        // C layout: VGPR r -> row r (+8 hi half)
            float v = acc[j][r] + bj;
            if (mode == 3) {
                reinterpret_cast<float*>(outp)[(size_t)m * 1024 + col] = v;
            } else {
                const int b = m >> 11, s = m & 2047;
                const int h = col >> 6, dd = col & 63;
                __bf16* o = reinterpret_cast<__bf16*>(outp);
                if (mode == 0)
                    o[((size_t)(b * 16 + h) * 2048 + s) * 64 + dd] = f2bf(v * 0.125f);
                else if (mode == 1)
                    o[((size_t)(b * 16 + h) * 2048 + s) * 64 + dd] = f2bf(v);
                else
                    o[((size_t)(b * 16 + h) * 64 + dd) * 2048 + s] = f2bf(v);
            }
        }
    }
}

// ---------------------------------------------------------------------------
// Kernel 4: banded flash attention. One wave per 16-row query tile.
// Scores via 2 WMMA (K=64 split in 2x32), streaming softmax over 32-key blocks,
// P transposed through per-wave LDS, P@V via 4 WMMA per block.
// ---------------------------------------------------------------------------
__global__ void lf_attn(const __bf16* __restrict__ Q, const __bf16* __restrict__ K,
                        const __bf16* __restrict__ Vt, __bf16* __restrict__ attnC) {
    __shared__ __bf16 ldsP[4][16 * 32];   // per-wave 16x32 P staging tile

    const int lane = threadIdx.x & 31;
    const int w    = threadIdx.x >> 5;
    const int gw   = blockIdx.x * 4 + w;    // 0..4095
    const int qt   = gw & 127;              // query tile within (b,h)
    const int bh   = gw >> 7;               // 0..31
    const int q0   = qt << 4;
    const int ml   = lane & 15;
    const int kh   = lane >> 4;
    __bf16* P = ldsP[w];

    // Q fragments for this tile (d=64 -> two 16x32 A frags), scale pre-folded.
    const __bf16* qrow = Q + ((size_t)bh * 2048 + q0 + ml) * 64;
    AFrag aQ0, aQ1;
    aQ0.p.lo = *reinterpret_cast<const v8bf*>(qrow + kh * 16);
    aQ0.p.hi = *reinterpret_cast<const v8bf*>(qrow + kh * 16 + 8);
    aQ1.p.lo = *reinterpret_cast<const v8bf*>(qrow + 32 + kh * 16);
    aQ1.p.hi = *reinterpret_cast<const v8bf*>(qrow + 32 + kh * 16 + 8);

    const v8f zf = {0.f, 0.f, 0.f, 0.f, 0.f, 0.f, 0.f, 0.f};
    v8f O[4] = {zf, zf, zf, zf};
    float rmax[8], rsum[8];
#pragma unroll
    for (int r = 0; r < 8; ++r) { rmax[r] = -1e30f; rsum[r] = 0.f; }

    auto block = [&](int j0) {
        // K fragments: K row-major [S,d] is exactly Bt[N=key][K=d].
        const __bf16* kb0 = K + ((size_t)bh * 2048 + j0 + ml) * 64;
        const __bf16* kb1 = kb0 + 16 * 64;
        const v16bf k00 = *reinterpret_cast<const v16bf*>(kb0 + kh * 16);
        const v16bf k01 = *reinterpret_cast<const v16bf*>(kb0 + 32 + kh * 16);
        const v16bf k10 = *reinterpret_cast<const v16bf*>(kb1 + kh * 16);
        const v16bf k11 = *reinterpret_cast<const v16bf*>(kb1 + 32 + kh * 16);

        v8f s0 = wmma_bf16(aQ0.v, k00, zf);
        s0     = wmma_bf16(aQ1.v, k01, s0);
        v8f s1 = wmma_bf16(aQ0.v, k10, zf);
        s1     = wmma_bf16(aQ1.v, k11, s1);

#pragma unroll
        for (int r = 0; r < 8; ++r) {
            const int row = q0 + kh * 8 + r;
            const int c0 = j0 + ml, c1 = c0 + 16;
            float v0 = s0[r], v1 = s1[r];
            int d0 = row - c0; if (d0 < 0) d0 = -d0;
            int d1 = row - c1; if (d1 < 0) d1 = -d1;
            if (!(d0 <= 256 || row == 0 || c0 == 0)) v0 = -1e30f;
            if (!(d1 <= 256 || row == 0 || c1 == 0)) v1 = -1e30f;

            float mx = fmaxf(v0, v1);                 // row max across 16 lanes
            mx = fmaxf(mx, swz<1>(mx));
            mx = fmaxf(mx, swz<2>(mx));
            mx = fmaxf(mx, swz<4>(mx));
            mx = fmaxf(mx, swz<8>(mx));

            const float nm  = fmaxf(rmax[r], mx);
            const float fac = __expf(rmax[r] - nm);
            rmax[r] = nm;
            const float p0 = __expf(v0 - nm);
            const float p1 = __expf(v1 - nm);
            float ts = p0 + p1;
            ts += swz<1>(ts);
            ts += swz<2>(ts);
            ts += swz<4>(ts);
            ts += swz<8>(ts);
            rsum[r] = rsum[r] * fac + ts;
#pragma unroll
            for (int nc = 0; nc < 4; ++nc) O[nc][r] *= fac;

            const int li = (kh * 8 + r) * 32 + ml;    // P[row][col] in LDS
            P[li]      = f2bf(p0);
            P[li + 16] = f2bf(p1);
        }

        asm volatile("s_wait_dscnt 0x0" ::: "memory");   // LDS stores visible

        AFrag aP;                                        // P as 16x32 A fragment
        const __bf16* pr = P + ml * 32 + kh * 16;
        aP.p.lo = *reinterpret_cast<const v8bf*>(pr);
        aP.p.hi = *reinterpret_cast<const v8bf*>(pr + 8);

#pragma unroll
        for (int nc = 0; nc < 4; ++nc) {
            const v16bf vf = *reinterpret_cast<const v16bf*>(
                Vt + ((size_t)bh * 64 + nc * 16 + ml) * 2048 + j0 + kh * 16);
            O[nc] = wmma_bf16(aP.v, vf, O[nc]);
        }
    };

    if (qt == 0) {
        // tile contains global row 0 -> sweep all keys
        for (int j0 = 0; j0 < 2048; j0 += 32) block(j0);
    } else {
        block(0);                               // global column first (numerics)
        int lo = q0 - 256; if (lo < 0) lo = 0;
        lo &= ~31;
        int hi = q0 + 271; if (hi > 2047) hi = 2047;
        const int start = (lo == 0) ? 32 : lo;
        for (int j0 = start; j0 <= hi; j0 += 32) block(j0);
    }

    // Normalize and write merged-head bf16 [B*S, 1024] for the output GEMM.
    const int b = bh >> 4, h = bh & 15;
#pragma unroll
    for (int r = 0; r < 8; ++r) {
        const float inv = 1.0f / rsum[r];
        const int s = q0 + kh * 8 + r;
#pragma unroll
        for (int nc = 0; nc < 4; ++nc) {
            const int dd = nc * 16 + ml;
            attnC[((size_t)(b * 2048) + s) * 1024 + h * 64 + dd] = f2bf(O[nc][r] * inv);
        }
    }
}

// ---------------------------------------------------------------------------
// Host-side launch. Inputs (setup_inputs order):
//  0:x 1:Wq 2:bq 3:Wk 4:bk 5:Wv 6:bv 7:Wo 8:bo
// ---------------------------------------------------------------------------
extern "C" void kernel_launch(void* const* d_in, const int* in_sizes, int n_in,
                              void* d_out, int out_size, void* d_ws, size_t ws_size,
                              hipStream_t stream) {
    (void)in_sizes; (void)n_in; (void)out_size; (void)ws_size;

    const float* x  = (const float*)d_in[0];
    const float* Wq = (const float*)d_in[1]; const float* bq = (const float*)d_in[2];
    const float* Wk = (const float*)d_in[3]; const float* bk = (const float*)d_in[4];
    const float* Wv = (const float*)d_in[5]; const float* bv = (const float*)d_in[6];
    const float* Wo = (const float*)d_in[7]; const float* bo = (const float*)d_in[8];
    float* out = (float*)d_out;

    char* ws = (char*)d_ws;
    const size_t SZ_X  = (size_t)4096 * 1024 * 2;   // bf16
    const size_t SZ_W  = (size_t)1024 * 1024 * 2;   // bf16
    const size_t SZ_H  = (size_t)2 * 16 * 2048 * 64 * 2;

    __bf16* xb    = (__bf16*)(ws);                      ws += SZ_X;
    __bf16* Wtq   = (__bf16*)(ws);                      ws += SZ_W;
    __bf16* Wtk   = (__bf16*)(ws);                      ws += SZ_W;
    __bf16* Wtv   = (__bf16*)(ws);                      ws += SZ_W;
    __bf16* Wto   = (__bf16*)(ws);                      ws += SZ_W;
    __bf16* Qh    = (__bf16*)(ws);                      ws += SZ_H;
    __bf16* Kh    = (__bf16*)(ws);                      ws += SZ_H;
    __bf16* Vth   = (__bf16*)(ws);                      ws += SZ_H;
    __bf16* attnC = (__bf16*)(ws);

    lf_cvt_x      <<<16384, 256, 0, stream>>>(x, xb);
    lf_transpose_w<<< 4096, 256, 0, stream>>>(Wq, Wtq);
    lf_transpose_w<<< 4096, 256, 0, stream>>>(Wk, Wtk);
    lf_transpose_w<<< 4096, 256, 0, stream>>>(Wv, Wtv);
    lf_transpose_w<<< 4096, 256, 0, stream>>>(Wo, Wto);

    lf_gemm<<<512, 256, 0, stream>>>(xb, Wtq, bq, (void*)Qh,  0);
    lf_gemm<<<512, 256, 0, stream>>>(xb, Wtk, bk, (void*)Kh,  1);
    lf_gemm<<<512, 256, 0, stream>>>(xb, Wtv, bv, (void*)Vth, 2);

    lf_attn<<<1024, 128, 0, stream>>>(Qh, Kh, Vth, attnC);

    lf_gemm<<<512, 256, 0, stream>>>(attnC, Wto, bo, (void*)out, 3);
}
